// multi_GRU_68874095558834
// MI455X (gfx1250) — matrix operand order repaired and checked by reading, not compiled, not run on hardware
//
#include <hip/hip_runtime.h>
#include <hip/hip_bf16.h>
#include <cstddef>

typedef __attribute__((ext_vector_type(16))) __bf16 v16bf;
typedef __attribute__((ext_vector_type(8)))  __bf16 v8bf;
typedef __attribute__((ext_vector_type(8)))  float  v8f;
typedef __attribute__((ext_vector_type(4)))  float  v4f;

#define CH   128
#define HH_  96
#define WW_  96
#define BB_  16
#define KTOT 256          // Cin of every conv (concat of two 128-ch tensors)
#define KSTR 264          // LDS row stride in bf16 elements (padded, 16B aligned)

__device__ __forceinline__ float sig_f(float t)  { return 1.0f / (1.0f + __expf(-t)); }

// ---------------------------------------------------------------------------
// Weight prep: f32 OIHW [N,Cin,KH,KW] -> bf16 [N][tap][Cin] (tap = kh*KW+kw)
// Every B-fragment becomes a single contiguous 32-byte load per lane.
// ---------------------------------------------------------------------------
__global__ void wprep_kernel(const float* __restrict__ src, __bf16* __restrict__ dst,
                             int N, int Cin, int taps) {
    int i = blockIdx.x * blockDim.x + threadIdx.x;
    int tot = N * Cin * taps;
    if (i >= tot) return;
    int k  = i % Cin;
    int nt = i / Cin;
    int t  = nt % taps;
    int n  = nt / taps;
    dst[i] = (__bf16)src[((size_t)n * Cin + k) * taps + t];
}

// ---------------------------------------------------------------------------
// Fused conv-as-GEMM (bf16 WMMA, f32 accumulate).
//   TAPS : 1 (pointwise) or 3
//   DIR  : 0 = taps along w (1x3), 1 = taps along h (3x1)
//   EMODE: 0 : N=256. A/aux are f32 (x,y). n<128: h=acc+bd1 -> out0(bf16).
//              n>=128: p=sigmoid(acc+bd2); out1(bf16) = p*x + (1-p)*y
//          1 : N=256. A/aux bf16. n<128: out0(f32) = sigmoid(acc+b0)   (z)
//              n>=128: r=sigmoid(acc+b1); out1(bf16) = r*h
//          2 : N=128. q=tanh(acc+b0); z=aux0(f32); xv=aux1(bf16);
//              out0 = relu((1+z)*xv+(1-z)*q)  (bf16, or f32 if OUTF32)
// Block = 256 threads (8 waves). Grid = one 16-pixel M-tile per block.
// ---------------------------------------------------------------------------
template <int TAPS, int DIR, int EMODE, bool OUTF32>
__global__ __launch_bounds__(256)
void conv_gemm_kernel(const void* __restrict__ A0v, const void* __restrict__ A1v,
                      const __bf16* __restrict__ wbuf,
                      const float* __restrict__ bias0, const float* __restrict__ bias1,
                      const void* __restrict__ aux0v,  const void* __restrict__ aux1v,
                      void* __restrict__ out0v, void* __restrict__ out1v) {
    __shared__ __bf16 lds[TAPS * 16 * KSTR];

    const int tile = blockIdx.x;             // B*H*(W/16) tiles
    const int wb   = (tile % (WW_ / 16)) * 16;
    const int row  = tile / (WW_ / 16);      // b*H + h
    const int b    = row / HH_;
    const int hrow = row % HH_;
    const size_t plane = (size_t)HH_ * WW_;

    // ---- stage activation tile [TAPS][16 pixels][256 ch] into LDS as bf16 ----
    for (int i = threadIdx.x; i < TAPS * 16 * KTOT; i += 256) {
        int m = i & 15;
        int t = (i >> 4) % TAPS;
        int k = i / (16 * TAPS);
        int hp = hrow + ((DIR == 1) ? (t - TAPS / 2) : 0);
        int wp = wb + m + ((DIR == 0) ? (t - TAPS / 2) : 0);
        __bf16 v = (__bf16)0.0f;
        if (hp >= 0 && hp < HH_ && wp >= 0 && wp < WW_) {
            size_t idx = ((size_t)b * CH + (k & (CH - 1))) * plane
                       + (size_t)hp * WW_ + wp;
            if constexpr (EMODE == 0) {
                const float* s = (k < CH) ? (const float*)A0v : (const float*)A1v;
                v = (__bf16)s[idx];
            } else {
                const __bf16* s = (k < CH) ? (const __bf16*)A0v : (const __bf16*)A1v;
                v = s[idx];
            }
        }
        lds[(t * 16 + m) * KSTR + k] = v;
    }
    __syncthreads();

    const int wave  = threadIdx.x >> 5;
    const int lane  = threadIdx.x & 31;
    const int lhalf = lane >> 4;      // 0: lanes 0-15, 1: lanes 16-31
    const int ln    = lane & 15;
    const int ntiles = (EMODE == 2) ? 8 : 16;

    for (int nt = wave; nt < ntiles; nt += 8) {
        const int n = nt * 16 + ln;   // output channel owned by this lane's column
        __builtin_prefetch(&wbuf[((size_t)n * TAPS) * KTOT], 0, 3);

        v8f acc = {};
#pragma unroll
        for (int t = 0; t < TAPS; ++t) {
            // Load all 8 B-fragments of this tap into distinct registers first:
            // one multi-load clause, staggered s_wait, WMMAs overlap the loads.
            v16bf bfr[8];
#pragma unroll
            for (int kb = 0; kb < 8; ++kb)
                bfr[kb] = *(const v16bf*)
                    &wbuf[((size_t)n * TAPS + t) * KTOT + kb * 32 + lhalf * 16];
#pragma unroll
            for (int kb = 0; kb < 8; ++kb) {
                // A fragment: lane<16 -> K {0..15}; lane>=16 -> {16..31}
                const __bf16* ap = &lds[(t * 16 + ln) * KSTR + kb * 32 + lhalf * 16];
                v8bf alo = *(const v8bf*)ap;
                v8bf ahi = *(const v8bf*)(ap + 8);
                v16bf af = __builtin_shufflevector(alo, ahi,
                        0, 1, 2, 3, 4, 5, 6, 7, 8, 9, 10, 11, 12, 13, 14, 15);
                acc = __builtin_amdgcn_wmma_f32_16x16x32_bf16(
                        false, af, false, bfr[kb], (short)0, acc, false, false);
            }
        }

        // ---- epilogue: lane holds column n, rows m0..m0+7 (consecutive w) ----
        const int m0 = lhalf * 8;
        const size_t rowoff = (size_t)hrow * WW_ + wb + m0;
        const int c = n & (CH - 1);
        const size_t o = ((size_t)b * CH + c) * plane + rowoff;   // 16B aligned

        if constexpr (EMODE == 0) {
            if (n < CH) {                          // h = conv + bd1 -> bf16
                float bia = bias0[n];
                v8bf hv;
#pragma unroll
                for (int v = 0; v < 8; ++v) hv[v] = (__bf16)(acc[v] + bia);
                *(v8bf*)((__bf16*)out0v + o) = hv;
            } else {                               // p -> blend x,y -> bf16
                float bia = bias1[c];
                v4f x0 = *(const v4f*)((const float*)aux0v + o);
                v4f x1 = *(const v4f*)((const float*)aux0v + o + 4);
                v4f y0 = *(const v4f*)((const float*)aux1v + o);
                v4f y1 = *(const v4f*)((const float*)aux1v + o + 4);
                v8bf xb;
#pragma unroll
                for (int v = 0; v < 8; ++v) {
                    float p  = sig_f(acc[v] + bia);
                    float xv = (v < 4) ? x0[v & 3] : x1[v & 3];
                    float yv = (v < 4) ? y0[v & 3] : y1[v & 3];
                    xb[v] = (__bf16)(p * xv + (1.0f - p) * yv);
                }
                *(v8bf*)((__bf16*)out1v + o) = xb;
            }
        } else if constexpr (EMODE == 1) {
            if (n < CH) {                          // z -> f32
                float bia = bias0[n];
                v4f z0, z1;
#pragma unroll
                for (int v = 0; v < 8; ++v) {
                    float z = sig_f(acc[v] + bia);
                    if (v < 4) z0[v & 3] = z; else z1[v & 3] = z;
                }
                *(v4f*)((float*)out0v + o)     = z0;
                *(v4f*)((float*)out0v + o + 4) = z1;
            } else {                               // r*h -> bf16
                float bia = bias1[c];
                v8bf hv = *(const v8bf*)((const __bf16*)aux0v + o);
                v8bf rh;
#pragma unroll
                for (int v = 0; v < 8; ++v) {
                    float r = sig_f(acc[v] + bia);
                    rh[v] = (__bf16)(r * (float)hv[v]);
                }
                *(v8bf*)((__bf16*)out1v + o) = rh;
            }
        } else {                                   // q + GRU update
            float bia = bias0[n];
            v4f z0 = *(const v4f*)((const float*)aux0v + o);
            v4f z1 = *(const v4f*)((const float*)aux0v + o + 4);
            v8bf xv = *(const v8bf*)((const __bf16*)aux1v + o);
            if constexpr (OUTF32) {
                v4f r0, r1;
#pragma unroll
                for (int v = 0; v < 8; ++v) {
                    float q = tanhf(acc[v] + bia);
                    float z = (v < 4) ? z0[v & 3] : z1[v & 3];
                    float r = (1.0f + z) * (float)xv[v] + (1.0f - z) * q;
                    r = r > 0.0f ? r : 0.0f;
                    if (v < 4) r0[v & 3] = r; else r1[v & 3] = r;
                }
                *(v4f*)((float*)out0v + o)     = r0;
                *(v4f*)((float*)out0v + o + 4) = r1;
            } else {
                v8bf rr;
#pragma unroll
                for (int v = 0; v < 8; ++v) {
                    float q = tanhf(acc[v] + bia);
                    float z = (v < 4) ? z0[v & 3] : z1[v & 3];
                    float r = (1.0f + z) * (float)xv[v] + (1.0f - z) * q;
                    rr[v] = (__bf16)(r > 0.0f ? r : 0.0f);
                }
                *(v8bf*)((__bf16*)out0v + o) = rr;
            }
        }
    }
}

// ---------------------------------------------------------------------------
extern "C" void kernel_launch(void* const* d_in, const int* in_sizes, int n_in,
                              void* d_out, int out_size, void* d_ws, size_t ws_size,
                              hipStream_t stream) {
    const float* x   = (const float*)d_in[0];
    const float* y   = (const float*)d_in[1];
    const float* Wd1 = (const float*)d_in[2];  const float* bd1 = (const float*)d_in[3];
    const float* Wd2 = (const float*)d_in[4];  const float* bd2 = (const float*)d_in[5];
    const float* Wz1 = (const float*)d_in[6];  const float* bz1 = (const float*)d_in[7];
    const float* Wr1 = (const float*)d_in[8];  const float* br1 = (const float*)d_in[9];
    const float* Wq1 = (const float*)d_in[10]; const float* bq1 = (const float*)d_in[11];
    const float* Wz2 = (const float*)d_in[12]; const float* bz2 = (const float*)d_in[13];
    const float* Wr2 = (const float*)d_in[14]; const float* br2 = (const float*)d_in[15];
    const float* Wq2 = (const float*)d_in[16]; const float* bq2 = (const float*)d_in[17];

    // ---- workspace layout ----
    __bf16* wbp = (__bf16*)d_ws;
    const size_t W11   = (size_t)CH * KTOT;       // 32768 elems (1x1 weight)
    const size_t W13   = (size_t)CH * KTOT * 3;   // 98304 elems (1x3 / 3x1 weight)
    __bf16* wd12 = wbp;                           // Wd1 | Wd2  (n = 0..255)
    __bf16* wzr1 = wbp + 2 * W11;                 // Wz1 | Wr1
    __bf16* wq1  = wzr1 + 2 * W13;
    __bf16* wzr2 = wq1 + W13;                     // Wz2 | Wr2
    __bf16* wq2  = wzr2 + 2 * W13;
    size_t wtot_bytes = (2 * W11 + 6 * W13) * sizeof(__bf16);   // 1,310,720 B

    const size_t NT = (size_t)BB_ * CH * HH_ * WW_;             // 18,874,368 elems
    float*  zbuf  = (float*)((char*)d_ws + wtot_bytes);         // z, f32
    __bf16* hbuf  = (__bf16*)(zbuf + NT);                       // h
    __bf16* xbbuf = hbuf + NT;                                  // blended x
    __bf16* rhbuf = xbbuf + NT;                                 // r*h
    __bf16* x1buf = rhbuf + NT;                                 // x after stage 1
    float*  xout  = (float*)d_out;

    // ---- pack weights to bf16, [n][tap][k] ----
    {
        dim3 blk(256);
        int t1 = CH * KTOT, t3 = CH * KTOT * 3;
        dim3 g1((t1 + 255) / 256), g3((t3 + 255) / 256);
        wprep_kernel<<<g1, blk, 0, stream>>>(Wd1, wd12,        CH, KTOT, 1);
        wprep_kernel<<<g1, blk, 0, stream>>>(Wd2, wd12 + W11,  CH, KTOT, 1);
        wprep_kernel<<<g3, blk, 0, stream>>>(Wz1, wzr1,        CH, KTOT, 3);
        wprep_kernel<<<g3, blk, 0, stream>>>(Wr1, wzr1 + W13,  CH, KTOT, 3);
        wprep_kernel<<<g3, blk, 0, stream>>>(Wq1, wq1,         CH, KTOT, 3);
        wprep_kernel<<<g3, blk, 0, stream>>>(Wz2, wzr2,        CH, KTOT, 3);
        wprep_kernel<<<g3, blk, 0, stream>>>(Wr2, wzr2 + W13,  CH, KTOT, 3);
        wprep_kernel<<<g3, blk, 0, stream>>>(Wq2, wq2,         CH, KTOT, 3);
    }

    dim3 grid(BB_ * HH_ * (WW_ / 16));   // 9216 M-tiles
    dim3 blk(256);

    // K1: h = conv1x1([x,y])+bd1 ; p = sigmoid(conv1x1+bd2) ; xb = p*x+(1-p)*y
    conv_gemm_kernel<1, 0, 0, false><<<grid, blk, 0, stream>>>(
        x, y, wd12, bd1, bd2, x, y, hbuf, xbbuf);

    // Stage 1 (horizontal 1x3)
    conv_gemm_kernel<3, 0, 1, false><<<grid, blk, 0, stream>>>(
        hbuf, xbbuf, wzr1, bz1, br1, hbuf, nullptr, zbuf, rhbuf);
    conv_gemm_kernel<3, 0, 2, false><<<grid, blk, 0, stream>>>(
        rhbuf, xbbuf, wq1, bq1, nullptr, zbuf, xbbuf, x1buf, nullptr);

    // Stage 2 (vertical 3x1)
    conv_gemm_kernel<3, 1, 1, false><<<grid, blk, 0, stream>>>(
        hbuf, x1buf, wzr2, bz2, br2, hbuf, nullptr, zbuf, rhbuf);
    conv_gemm_kernel<3, 1, 2, true><<<grid, blk, 0, stream>>>(
        rhbuf, x1buf, wq2, bq2, nullptr, zbuf, x1buf, xout, nullptr);
}
